// CustomLSTM_With_Peephole_45801531244559
// MI455X (gfx1250) — compile-verified, matrix operand order checked by reading, not run on hardware
//
#include <hip/hip_runtime.h>
#include <math.h>

typedef __attribute__((ext_vector_type(16))) __bf16 v16bf;
typedef __attribute__((ext_vector_type(8)))  float  v8f;

static constexpr int Bb  = 64;    // batch
static constexpr int Ss  = 512;   // sequence
static constexpr int Din = 256;   // input dim
static constexpr int Hh  = 1024;  // hidden
static constexpr int NT  = 4 * Hh;                       // packed gate columns (j*4+g)
static constexpr size_t OUT_BSTRIDE = (size_t)Ss * 2 * Hh; // batch stride in d_out

// CDNA5 async LDS fill: 16 bytes per lane, tracked by ASYNCcnt (no VGPR data).
__device__ __forceinline__ void async_b128(unsigned lds_byte_off, const void* gaddr)
{
  asm volatile("global_load_async_to_lds_b128 %0, %1, off"
               :: "v"(lds_byte_off), "v"(gaddr) : "memory");
}

// ---------------------------------------------------------------------------
// Step kernel: one LSTM timestep for both directions (blockIdx.y = dir).
// pre = [x_t ; h_{t-1}] @ Wt  (K = Kx + H), Wt packed [NT][Ktot] bf16 with
// n = 4*j + gate (f,i,c,o), then the peephole LSTM cell update.
// Each block owns a 64(batch) x 64(packed cols) tile = all 4 gates of 16 units.
// When hprev == null (t == 0) the h-segment is skipped entirely (h == 0).
// ---------------------------------------------------------------------------
struct DirArgs {
  const __bf16* xt;       // [B][Kx] layer input at this timestep
  const __bf16* hprev;    // [B][H] previous hidden (bf16), null at first step
  const __bf16* wt;       // [NT][Ktot] packed weights
  const float*  bias;     // [NT]
  const float*  peep;     // [NT] (0 for gate c)
  float*        c;        // [B][H] cell state (fp32)
  __bf16*       hout_bf;  // [B][H] hidden out (bf16, feeds recurrence / next layer)
  float*        hout_f32; // into d_out (pre-offset), batch stride OUT_BSTRIDE; may be null
};
struct StepArgs { DirArgs d[2]; int Kx; };

__global__ __launch_bounds__(256) void lstm_step(StepArgs sa)
{
  const DirArgs a  = sa.d[blockIdx.y];
  const int Kx     = sa.Kx;
  const int Ktot   = Kx + Hh;
  const int n0     = blockIdx.x * 64;

  __shared__ __align__(16) __bf16 As[2][64 * 40]; // double-buffered activation panel
  __shared__ __align__(16) __bf16 Bs[2][64 * 40]; // double-buffered weight panel
  __shared__ float P[64 * 65];                    // pre-activations

  const int tid  = threadIdx.x;
  const int lane = tid & 31;
  const int wave = tid >> 5;
  const int mt   = wave & 3;          // m-tile (16 rows)
  const int ntb  = (wave >> 2) * 2;   // first of two n-tiles

  const int sm = tid >> 2;            // staging row 0..63
  const int se = (tid & 3) * 8;       // staging element offset 0,8,16,24

  // per-thread LDS staging destinations (byte offsets within LDS)
  const unsigned ldsA[2] = { (unsigned)(size_t)&As[0][sm * 40 + se],
                             (unsigned)(size_t)&As[1][sm * 40 + se] };
  const unsigned ldsB[2] = { (unsigned)(size_t)&Bs[0][sm * 40 + se],
                             (unsigned)(size_t)&Bs[1][sm * 40 + se] };

  // fragment gather offsets (elements)
  const int hr  = lane & 15;
  const int kh  = (lane >> 4) * 8;    // lanes 16-31 hold K+8
  const int ma  = (mt * 16 + hr) * 40 + kh;
  const int na0 = ((ntb)     * 16 + hr) * 40 + kh;
  const int na1 = ((ntb + 1) * 16 + hr) * 40 + kh;

  const __bf16* wrow = a.wt + (size_t)(n0 + sm) * Ktot + se; // weight staging row

  v8f acc0 = {}; v8f acc1 = {};

  // One GEMM segment: nkb K-blocks of 32, activations from `act` (row stride ldA),
  // weights from column offset kw0. Async double-buffered pipeline.
  auto run_seg = [&](const __bf16* act, int ldA, int kw0, int nkb) {
    const __bf16* arow = act + (size_t)sm * ldA + se;
    const __bf16* brow = wrow + kw0;
    async_b128(ldsA[0], arow);
    async_b128(ldsB[0], brow);
    for (int i = 0; i < nkb; ++i) {
      const int cur = i & 1;
      if (i + 1 < nkb) {
        async_b128(ldsA[cur ^ 1], arow + (size_t)(i + 1) * 32);
        async_b128(ldsB[cur ^ 1], brow + (size_t)(i + 1) * 32);
        asm volatile("s_wait_asynccnt 0x2" ::: "memory");  // current buffer landed
      } else {
        asm volatile("s_wait_asynccnt 0x0" ::: "memory");
      }
      __syncthreads();
      union Frag { v16bf v; uint4 q[2]; };
      Frag af, bf0, bf1;
      const __bf16* Ab = As[cur];
      const __bf16* Bp = Bs[cur];
      af.q[0]  = *(const uint4*)&Ab[ma];        af.q[1]  = *(const uint4*)&Ab[ma + 16];
      bf0.q[0] = *(const uint4*)&Bp[na0];       bf0.q[1] = *(const uint4*)&Bp[na0 + 16];
      bf1.q[0] = *(const uint4*)&Bp[na1];       bf1.q[1] = *(const uint4*)&Bp[na1 + 16];
      acc0 = __builtin_amdgcn_wmma_f32_16x16x32_bf16(false, af.v, false, bf0.v,
                                                     (short)0, acc0, false, false);
      acc1 = __builtin_amdgcn_wmma_f32_16x16x32_bf16(false, af.v, false, bf1.v,
                                                     (short)0, acc1, false, false);
      __syncthreads();  // all waves done reading buf[cur] before it is refilled
    }
  };

  run_seg(a.xt, Kx, 0, Kx / 32);
  if (a.hprev) run_seg(a.hprev, Hh, Kx, Hh / 32);

  // ---- spill pre-activations to LDS (D layout: m = r + (lane>>4)*8, n = lane&15) ----
  {
    const int mh = (lane >> 4) * 8;
    #pragma unroll
    for (int r = 0; r < 8; ++r) {
      P[(mt * 16 + r + mh) * 65 + (ntb)     * 16 + hr] = acc0[r];
      P[(mt * 16 + r + mh) * 65 + (ntb + 1) * 16 + hr] = acc1[r];
    }
  }
  __syncthreads();

  // ---- elementwise peephole LSTM update: each thread owns 4 (m, j) pairs ----
  const float* bias = a.bias + n0;
  const float* peep = a.peep + n0;
  #pragma unroll
  for (int p = 0; p < 4; ++p) {
    const int idx = tid + p * 256;
    const int m   = idx >> 4;
    const int jl  = idx & 15;
    const int jg  = (n0 >> 2) + jl;
    const float cold = a.c[m * Hh + jg];
    const float pf = P[m * 65 + jl * 4 + 0] + bias[jl * 4 + 0] + peep[jl * 4 + 0] * cold;
    const float pi = P[m * 65 + jl * 4 + 1] + bias[jl * 4 + 1] + peep[jl * 4 + 1] * cold;
    const float pc = P[m * 65 + jl * 4 + 2] + bias[jl * 4 + 2];
    const float fg = 1.0f / (1.0f + __expf(-pf));
    const float ig = 1.0f / (1.0f + __expf(-pi));
    const float gg = tanhf(pc);
    const float cn = fg * cold + ig * gg;
    const float po = P[m * 65 + jl * 4 + 3] + bias[jl * 4 + 3] + peep[jl * 4 + 3] * cn;
    const float og = 1.0f / (1.0f + __expf(-po));
    const float hv = og * tanhf(cn);
    a.c[m * Hh + jg]       = cn;
    a.hout_bf[m * Hh + jg] = (__bf16)hv;
    if (a.hout_f32) a.hout_f32[(size_t)m * OUT_BSTRIDE + jg] = hv;
  }
}

// ---------------------------------------------------------------------------
// Packing kernels
// ---------------------------------------------------------------------------
struct PackWArgs { const float* U[4]; const float* W[4]; __bf16* dst; int Kx; int Ktot; };
__global__ void pack_w(PackWArgs pa)
{
  const size_t id = (size_t)blockIdx.x * 256 + threadIdx.x;  // n*Ktot + k
  const int k = (int)(id % (size_t)pa.Ktot);
  const int n = (int)(id / (size_t)pa.Ktot);
  const int g = n & 3, j = n >> 2;
  const float v = (k < pa.Kx) ? pa.U[g][(size_t)k * Hh + j]
                              : pa.W[g][(size_t)(k - pa.Kx) * Hh + j];
  pa.dst[id] = (__bf16)v;
}

struct PackBArgs { const float* b[4]; const float* V[4]; float* bias; float* peep; };
__global__ void pack_b(PackBArgs pa)
{
  const int n = blockIdx.x * 256 + threadIdx.x;
  const int g = n & 3, j = n >> 2;
  pa.bias[n] = pa.b[g][j];
  pa.peep[n] = pa.V[g] ? pa.V[g][j] : 0.0f;
}

// x[B][S][IN] fp32 -> xb[S][B][IN] bf16
__global__ void cvt_x(const float* __restrict__ x, __bf16* __restrict__ xb)
{
  const size_t id = (size_t)blockIdx.x * 256 + threadIdx.x;
  const int k = (int)(id % Din);
  const size_t rem = id / Din;
  const int t = (int)(rem % Ss);
  const int b = (int)(rem / Ss);
  xb[((size_t)t * Bb + b) * Din + k] = (__bf16)x[id];
}

// ---------------------------------------------------------------------------
// Host launcher
// ---------------------------------------------------------------------------
struct Cell { const float* W[4]; const float* U[4]; const float* V[4]; const float* b[4]; };

extern "C" void kernel_launch(void* const* d_in, const int* in_sizes, int n_in,
                              void* d_out, int out_size, void* d_ws, size_t ws_size,
                              hipStream_t stream)
{
  (void)n_in; (void)out_size; (void)ws_size;

  // ---- detect flattening scheme (dict insertion order vs pytree-sorted) ----
  const bool xFirst  = (in_sizes[0] == Bb * Ss * Din);
  const int  cell0   = xFirst ? 1 : 0;
  const bool cellIns = (in_sizes[cell0] == Hh * Hh);  // W_f first => insertion order
  const int  xIdx    = xFirst ? 0 : 60;

  int base[2][2]; // [dir][layer], dir0 = fwd
  if (xFirst) { base[0][0] = 1;  base[0][1] = 16; base[1][0] = 31; base[1][1] = 46; }
  else        { base[1][0] = 0;  base[1][1] = 15; base[0][0] = 30; base[0][1] = 45; }

  int offW[4], offU[4], offV[4], offb[4]; // gate order f,i,c,o
  if (cellIns) {
    const int w[4] = {0,4,8,11}, u[4] = {1,5,9,12}, v[4] = {2,6,-1,13}, bo[4] = {3,7,10,14};
    for (int g = 0; g < 4; ++g) { offW[g]=w[g]; offU[g]=u[g]; offV[g]=v[g]; offb[g]=bo[g]; }
  } else {
    const int w[4] = {8,9,7,10}, u[4] = {1,2,0,3}, v[4] = {4,5,-1,6}, bo[4] = {12,13,11,14};
    for (int g = 0; g < 4; ++g) { offW[g]=w[g]; offU[g]=u[g]; offV[g]=v[g]; offb[g]=bo[g]; }
  }

  Cell cells[2][2];
  for (int d = 0; d < 2; ++d)
    for (int l = 0; l < 2; ++l) {
      const int bi = base[d][l];
      for (int g = 0; g < 4; ++g) {
        cells[d][l].W[g] = (const float*)d_in[bi + offW[g]];
        cells[d][l].U[g] = (const float*)d_in[bi + offU[g]];
        cells[d][l].V[g] = (offV[g] < 0) ? nullptr : (const float*)d_in[bi + offV[g]];
        cells[d][l].b[g] = (const float*)d_in[bi + offb[g]];
      }
    }
  const float* x = (const float*)d_in[xIdx];
  float* out = (float*)d_out;

  // ---- workspace suballocation (256B aligned) ----
  char* wp = (char*)d_ws;
  auto alloc = [&](size_t bytes) -> char* {
    char* p = wp; wp += (bytes + 255) & ~(size_t)255; return p;
  };
  __bf16* xb = (__bf16*)alloc((size_t)Ss * Bb * Din * 2);
  __bf16* hs0[2]; for (int d = 0; d < 2; ++d) hs0[d] = (__bf16*)alloc((size_t)Ss * Bb * Hh * 2);
  __bf16* h1[2];  for (int d = 0; d < 2; ++d) h1[d]  = (__bf16*)alloc((size_t)2 * Bb * Hh * 2);
  __bf16* wt[2][2];
  float *bias[2][2], *peep[2][2];
  const int KxL[2] = { Din, Hh };
  for (int d = 0; d < 2; ++d)
    for (int l = 0; l < 2; ++l) {
      const int Ktot = KxL[l] + Hh;
      wt[d][l]   = (__bf16*)alloc((size_t)NT * Ktot * 2);
      bias[d][l] = (float*)alloc((size_t)NT * 4);
      peep[d][l] = (float*)alloc((size_t)NT * 4);
    }
  float* cst = (float*)alloc((size_t)2 * Bb * Hh * 4);  // c state, both dirs

  // ---- 1) convert/transpose x ----
  cvt_x<<<(Bb * Ss * Din) / 256, 256, 0, stream>>>(x, xb);

  // ---- 2) pack weights / biases ----
  for (int d = 0; d < 2; ++d)
    for (int l = 0; l < 2; ++l) {
      PackWArgs pw;
      for (int g = 0; g < 4; ++g) { pw.U[g] = cells[d][l].U[g]; pw.W[g] = cells[d][l].W[g]; }
      pw.dst = wt[d][l]; pw.Kx = KxL[l]; pw.Ktot = KxL[l] + Hh;
      pack_w<<<(int)(((size_t)NT * pw.Ktot) / 256), 256, 0, stream>>>(pw);
      PackBArgs pb;
      for (int g = 0; g < 4; ++g) { pb.b[g] = cells[d][l].b[g]; pb.V[g] = cells[d][l].V[g]; }
      pb.bias = bias[d][l]; pb.peep = peep[d][l];
      pack_b<<<NT / 256, 256, 0, stream>>>(pb);
    }

  // ---- 3) layer 0 (fwd + bwd), 512 sequential steps ----
  hipMemsetAsync(cst, 0, (size_t)2 * Bb * Hh * 4, stream);
  for (int s = 0; s < Ss; ++s) {
    const int t0 = s, t1 = Ss - 1 - s;
    StepArgs sa; sa.Kx = Din;
    sa.d[0].xt = xb + (size_t)t0 * Bb * Din;
    sa.d[0].hprev = s ? (hs0[0] + (size_t)(t0 - 1) * Bb * Hh) : nullptr;
    sa.d[0].wt = wt[0][0]; sa.d[0].bias = bias[0][0]; sa.d[0].peep = peep[0][0];
    sa.d[0].c = cst;
    sa.d[0].hout_bf = hs0[0] + (size_t)t0 * Bb * Hh;
    sa.d[0].hout_f32 = nullptr;

    sa.d[1].xt = xb + (size_t)t1 * Bb * Din;
    sa.d[1].hprev = s ? (hs0[1] + (size_t)(t1 + 1) * Bb * Hh) : nullptr;
    sa.d[1].wt = wt[1][0]; sa.d[1].bias = bias[1][0]; sa.d[1].peep = peep[1][0];
    sa.d[1].c = cst + (size_t)Bb * Hh;
    sa.d[1].hout_bf = hs0[1] + (size_t)t1 * Bb * Hh;
    sa.d[1].hout_f32 = nullptr;

    lstm_step<<<dim3(64, 2), 256, 0, stream>>>(sa);
  }

  // ---- 4) layer 1 (fwd + bwd), writes d_out directly ----
  hipMemsetAsync(cst, 0, (size_t)2 * Bb * Hh * 4, stream);
  for (int s = 0; s < Ss; ++s) {
    const int t0 = s, t1 = Ss - 1 - s;
    const int cur = s & 1, prv = cur ^ 1;
    StepArgs sa; sa.Kx = Hh;
    sa.d[0].xt = hs0[0] + (size_t)t0 * Bb * Hh;
    sa.d[0].hprev = s ? (h1[0] + (size_t)prv * Bb * Hh) : nullptr;
    sa.d[0].wt = wt[0][1]; sa.d[0].bias = bias[0][1]; sa.d[0].peep = peep[0][1];
    sa.d[0].c = cst;
    sa.d[0].hout_bf = h1[0] + (size_t)cur * Bb * Hh;
    sa.d[0].hout_f32 = out + (size_t)t0 * 2 * Hh;           // out[:, t, 0:H]

    sa.d[1].xt = hs0[1] + (size_t)t1 * Bb * Hh;
    sa.d[1].hprev = s ? (h1[1] + (size_t)prv * Bb * Hh) : nullptr;
    sa.d[1].wt = wt[1][1]; sa.d[1].bias = bias[1][1]; sa.d[1].peep = peep[1][1];
    sa.d[1].c = cst + (size_t)Bb * Hh;
    sa.d[1].hout_bf = h1[1] + (size_t)cur * Bb * Hh;
    sa.d[1].hout_f32 = out + (size_t)s * 2 * Hh + Hh;       // out[:, s, H:2H] = b[::-1]

    lstm_step<<<dim3(64, 2), 256, 0, stream>>>(sa);
  }
}
